// WindowAttention_43173011259855
// MI455X (gfx1250) — compile-verified, matrix operand order checked by reading, not run on hardware
//
#include <hip/hip_runtime.h>
#include <hip/hip_bf16.h>

typedef __attribute__((ext_vector_type(16))) __bf16 bf16x16;
typedef __attribute__((ext_vector_type(8)))  float  f32x8;

// ---------- helpers ----------

__device__ __forceinline__ unsigned short f2bf(float f) {
    unsigned int u = __float_as_uint(f);
    unsigned int r = u + 0x7FFFu + ((u >> 16) & 1u);   // round-to-nearest-even
    return (unsigned short)(r >> 16);
}

__device__ __forceinline__ float bf2f(unsigned short u) {
    return __uint_as_float(((unsigned int)u) << 16);
}

union Frag {
    unsigned short u[16];
    bf16x16        v;
};

__device__ __forceinline__ f32x8 zero8() {
    f32x8 z;
#pragma unroll
    for (int i = 0; i < 8; ++i) z[i] = 0.0f;
    return z;
}

// A-operand (16x32, M x K) or B-operand stored as [N][K] rows, bf16 in LDS.
// CDNA5 16-bit A layout: lanes 0-15: u[0..7]=K0..7, u[8..15]=K16..23 (row=lane);
// lanes 16-31: u[0..7]=K8..15, u[8..15]=K24..31.
__device__ __forceinline__ void load_frag_rows(Frag& f, const unsigned short* T,
                                               int rowBase, int stride, int lane) {
    int r  = rowBase + (lane & 15);
    int kb = (lane >> 4) << 3;
    const unsigned short* p = T + r * stride + kb;
#pragma unroll
    for (int j = 0; j < 8; ++j) {
        f.u[j]     = p[j];
        f.u[8 + j] = p[16 + j];
    }
}

// B-operand stored as [K][N] (e.g. V matrix): column n = lane&15.
__device__ __forceinline__ void load_frag_kn(Frag& f, const unsigned short* T,
                                             int k0, int n0, int stride, int lane) {
    int n  = n0 + (lane & 15);
    int kb = k0 + ((lane >> 4) << 3);
#pragma unroll
    for (int j = 0; j < 8; ++j) {
        f.u[j]     = T[(kb + j) * stride + n];
        f.u[8 + j] = T[(kb + 16 + j) * stride + n];
    }
}

__device__ __forceinline__ f32x8 wmma_bf16(const Frag& a, const Frag& b, f32x8 c) {
    return __builtin_amdgcn_wmma_f32_16x16x32_bf16(false, a.v, false, b.v,
                                                   (short)0, c, false, false);
}

// gfx1250 async global->LDS (ASYNCcnt path). GVS mode: lds byte addr in VGPR,
// per-lane byte offset in VGPR, 64-bit base in SGPR pair. The immediate offset
// applies to BOTH the LDS and global addresses (ISA 10.x async pseudocode).
#define ASYNC_G2L_B128(ldsaddr, gbase, voff, imm)                               \
    asm volatile("global_load_async_to_lds_b128 %0, %1, %2 offset:" #imm        \
                 :: "v"(ldsaddr), "v"(voff), "s"(gbase) : "memory")

__device__ __forceinline__ void wait_async0() {
    asm volatile("s_wait_asynccnt 0x0" ::: "memory");
}
// Async loads complete in order: ASYNCcnt<=4 means the oldest 4 (current tile)
// have landed in LDS while the 4 prefetch loads for the next tile stay in flight.
__device__ __forceinline__ void wait_async4() {
    asm volatile("s_wait_asynccnt 0x4" ::: "memory");
}

__device__ __forceinline__ void stage_tile(unsigned lA, unsigned lB,
                                           const unsigned short* A,
                                           const unsigned short* B,
                                           int offA, int offB) {
    ASYNC_G2L_B128(lA, A, offA, 0);
    ASYNC_G2L_B128(lA, A, offA, 16);
    ASYNC_G2L_B128(lB, B, offB, 0);
    ASYNC_G2L_B128(lB, B, offB, 16);
}

// ---------- sizes ----------
#define Bw   2048      // windows
#define Nt   64        // tokens per window
#define Cd   512       // channels
#define Hh   16        // heads
#define Dd   32        // head dim
#define NW   64        // windows per image
#define ROWS (Bw * Nt) // 131072
#define KSTEPS (Cd / 32)

// ---------- kernel 1: fp32 -> bf16 convert ----------
__global__ void k_cvt(const float* __restrict__ src, unsigned short* __restrict__ dst,
                      size_t n) {
    size_t i      = (size_t)blockIdx.x * blockDim.x + threadIdx.x;
    size_t stride = (size_t)gridDim.x * blockDim.x;
    for (; i < n; i += stride) dst[i] = f2bf(src[i]);
}

// ---------- kernel 1b: fused qkv bias vector [1536] ----------
__global__ void k_qkvbias(const float* __restrict__ qb, const float* __restrict__ vb,
                          float* __restrict__ bias_full) {
    int o = blockIdx.x * blockDim.x + threadIdx.x;
    if (o >= 3 * Cd) return;
    float v = 0.f;
    if (o < Cd) v = qb[o];
    else if (o >= 2 * Cd) v = vb[o - 2 * Cd];
    bias_full[o] = v;
}

// ---------- kernel 2: CPB MLP -> bias_table [225][16] ----------
__global__ __launch_bounds__(256) void k_cpb(const float* __restrict__ rel_table,
                                             const float* __restrict__ w1,
                                             const float* __restrict__ b1,
                                             const float* __restrict__ w2,
                                             float* __restrict__ bias_table) {
    __shared__ float red[256];
    int   row = blockIdx.x;                 // 0..224
    float t0  = rel_table[row * 2 + 0];
    float t1  = rel_table[row * 2 + 1];
    int   tid = threadIdx.x;
    int   j0 = tid, j1 = tid + 256;
    float hid0 = fmaxf(0.f, t0 * w1[j0 * 2] + t1 * w1[j0 * 2 + 1] + b1[j0]);
    float hid1 = fmaxf(0.f, t0 * w1[j1 * 2] + t1 * w1[j1 * 2 + 1] + b1[j1]);
    for (int hd = 0; hd < 16; ++hd) {
        red[tid] = hid0 * w2[hd * 512 + j0] + hid1 * w2[hd * 512 + j1];
        __syncthreads();
        for (int s = 128; s > 0; s >>= 1) {
            if (tid < s) red[tid] += red[tid + s];
            __syncthreads();
        }
        if (tid == 0) bias_table[row * 16 + hd] = red[0];
        __syncthreads();
    }
}

// ---------- kernel 3: fused mask + 16*sigmoid(rpb) -> bc[64][16][64][64] ----------
__global__ void k_bias(const float* __restrict__ mask, const int* __restrict__ rel_index,
                       const float* __restrict__ bias_table, float* __restrict__ bc) {
    int idx = blockIdx.x * blockDim.x + threadIdx.x;
    if (idx >= NW * Hh * Nt * Nt) return;
    int nm   = idx & 4095;
    int rest = idx >> 12;
    int h    = rest & 15;
    int w    = rest >> 4;
    float rpb = bias_table[rel_index[nm] * 16 + h];
    float sig = 1.f / (1.f + __expf(-rpb));
    bc[idx]   = mask[w * 4096 + nm] + 16.f * sig;
}

// ---------- kernel 4: QKV GEMM (bf16 WMMA, double-buffered async LDS staging) ----
// scatter to bf16 qkv [3][B][H][N][d]
__global__ __launch_bounds__(128) void k_gemm_qkv(const unsigned short* __restrict__ Xb,
                                                  const unsigned short* __restrict__ Wb,
                                                  const float* __restrict__ bias_full,
                                                  unsigned short* __restrict__ qkvb) {
    __shared__ unsigned short As[2][64 * 32];
    __shared__ unsigned short Bs[2][64 * 32];
    const int m0   = blockIdx.x * 64;
    const int n0   = blockIdx.y * 64;
    const int tid  = threadIdx.x;
    const int wave = tid >> 5;
    const int lane = tid & 31;

    f32x8 acc[4];
#pragma unroll
    for (int i = 0; i < 4; ++i) acc[i] = zero8();

    const int r  = tid >> 1;
    const int hb = (tid & 1) * 16;
    const unsigned lA[2] = {(unsigned)(size_t)&As[0][r * 32 + hb],
                            (unsigned)(size_t)&As[1][r * 32 + hb]};
    const unsigned lB[2] = {(unsigned)(size_t)&Bs[0][r * 32 + hb],
                            (unsigned)(size_t)&Bs[1][r * 32 + hb]};
    const int rowA = ((m0 + r) * Cd + hb) * 2;   // bytes
    const int rowB = ((n0 + r) * Cd + hb) * 2;

    stage_tile(lA[0], lB[0], Xb, Wb, rowA, rowB);
    for (int i = 0; i < KSTEPS; ++i) {
        const int cur = i & 1;
        if (i + 1 < KSTEPS) {
            stage_tile(lA[cur ^ 1], lB[cur ^ 1], Xb, Wb,
                       rowA + (i + 1) * 64, rowB + (i + 1) * 64);
            wait_async4();
        } else {
            wait_async0();
        }
        __syncthreads();

        Frag a;
        load_frag_rows(a, As[cur], wave * 16, 32, lane);
#pragma unroll
        for (int nt = 0; nt < 4; ++nt) {
            Frag b;
            load_frag_rows(b, Bs[cur], nt * 16, 32, lane);
            acc[nt] = wmma_bf16(a, b, acc[nt]);
        }
        __syncthreads();   // all waves done reading buf[cur]; safe to refill next iter
    }

#pragma unroll
    for (int nt = 0; nt < 4; ++nt) {
#pragma unroll
        for (int rr = 0; rr < 8; ++rr) {
            int m  = wave * 16 + rr + ((lane >> 4) << 3);
            int o  = n0 + nt * 16 + (lane & 15);
            int gr = m0 + m;
            int b  = gr >> 6;
            int n  = gr & 63;
            int which = o >> 9;
            int oc    = o & 511;
            int hh    = oc >> 5;
            int dd    = oc & 31;
            qkvb[((((size_t)which * Bw + b) * Hh + hh) * Nt + n) * Dd + dd] =
                f2bf(acc[nt][rr] + bias_full[o]);
        }
    }
}

// ---------- kernel 5: per-(b,h) cosine attention (bf16 WMMA) ----------
__global__ __launch_bounds__(128) void k_attn(const unsigned short* __restrict__ qkvb,
                                              const float* __restrict__ logit_scale,
                                              const float* __restrict__ bc,
                                              unsigned short* __restrict__ ao) {
    __shared__ unsigned short qs[Nt * Dd];
    __shared__ unsigned short ks[Nt * Dd];
    __shared__ unsigned short vs[Nt * Dd];
    __shared__ float          ps[Nt][Nt];
    __shared__ unsigned short pb[Nt * Nt];

    const int b    = blockIdx.x;
    const int h    = blockIdx.y;
    const int tid  = threadIdx.x;
    const int wave = tid >> 5;
    const int lane = tid & 31;

    const size_t plane = (size_t)Bw * Hh * Nt * Dd;
    const size_t base  = (((size_t)b * Hh + h) * Nt) * Dd;
    const unsigned short* qg = qkvb + base;
    const unsigned short* kg = qkvb + plane + base;
    const unsigned short* vg = qkvb + 2 * plane + base;

    // V: pure copy -> async global->LDS, overlapped with q/k normalization below
    {
        const unsigned vlds = (unsigned)(size_t)&vs[tid * 16];
        const int      voff = tid * 32;   // bytes
        ASYNC_G2L_B128(vlds, vg, voff, 0);
        ASYNC_G2L_B128(vlds, vg, voff, 16);
    }

    // load + l2-normalize q (waves 0-1) / k (waves 2-3), bf16 into LDS
    {
        const unsigned short* src = (tid < 64) ? qg : kg;
        unsigned short*       dst = (tid < 64) ? qs : ks;
        int rrow = tid & 63;
        const unsigned short* row = src + rrow * Dd;
        float ss = 0.f;
        float tmp[Dd];
#pragma unroll
        for (int j = 0; j < Dd; ++j) { tmp[j] = bf2f(row[j]); ss += tmp[j] * tmp[j]; }
        float inv = 1.f / fmaxf(sqrtf(ss), 1e-12f);
#pragma unroll
        for (int j = 0; j < Dd; ++j) dst[rrow * Dd + j] = f2bf(tmp[j] * inv);
    }
    wait_async0();
    __syncthreads();

    const float scale = __expf(fminf(logit_scale[h], 4.60517018598809136804f)); // ln 100

    // QK^T: d=32 => exactly one 16x16x32 WMMA per tile
    const float* bch = bc + (((size_t)(b & (NW - 1)) * Hh + h) * Nt) * Nt;
    {
        Frag a;
        load_frag_rows(a, qs, wave * 16, Dd, lane);
#pragma unroll
        for (int nt = 0; nt < 4; ++nt) {
            Frag bk;
            load_frag_rows(bk, ks, nt * 16, Dd, lane);   // k row n is column n of K^T
            f32x8 acc = zero8();
            acc = wmma_bf16(a, bk, acc);
#pragma unroll
            for (int rr = 0; rr < 8; ++rr) {
                int m = wave * 16 + rr + ((lane >> 4) << 3);
                int n = nt * 16 + (lane & 15);
                ps[m][n] = acc[rr] * scale + bch[m * Nt + n];
            }
        }
    }
    __syncthreads();

    // row softmax, output bf16 probabilities
    if (tid < Nt) {
        float mx = -1e30f;
        for (int j = 0; j < Nt; ++j) mx = fmaxf(mx, ps[tid][j]);
        float s = 0.f;
        for (int j = 0; j < Nt; ++j) {
            float e = __expf(ps[tid][j] - mx);
            ps[tid][j] = e;
            s += e;
        }
        float inv = 1.f / s;
        for (int j = 0; j < Nt; ++j) pb[tid * Nt + j] = f2bf(ps[tid][j] * inv);
    }
    __syncthreads();

    // P @ V : K=64 (2 chunks), N=32 (2 tiles)
#pragma unroll
    for (int nt = 0; nt < 2; ++nt) {
        f32x8 acc = zero8();
#pragma unroll
        for (int kc = 0; kc < 2; ++kc) {
            Frag ap;
            load_frag_rows(ap, pb + kc * 32, wave * 16, Nt, lane);
            Frag bv;
            load_frag_kn(bv, vs, kc * 32, nt * 16, Dd, lane);
            acc = wmma_bf16(ap, bv, acc);
        }
#pragma unroll
        for (int rr = 0; rr < 8; ++rr) {
            int    m   = wave * 16 + rr + ((lane >> 4) << 3);
            int    col = h * Dd + nt * 16 + (lane & 15);
            size_t gr  = (size_t)b * Nt + m;
            ao[gr * Cd + col] = f2bf(acc[rr]);
        }
    }
}

// ---------- kernel 6: projection GEMM (bf16 WMMA, double-buffered async) -> fp32 ----
__global__ __launch_bounds__(128) void k_gemm_proj(const unsigned short* __restrict__ Ab,
                                                   const unsigned short* __restrict__ Wb,
                                                   const float* __restrict__ proj_b,
                                                   float* __restrict__ out) {
    __shared__ unsigned short As[2][64 * 32];
    __shared__ unsigned short Bs[2][64 * 32];
    const int m0   = blockIdx.x * 64;
    const int n0   = blockIdx.y * 64;
    const int tid  = threadIdx.x;
    const int wave = tid >> 5;
    const int lane = tid & 31;

    f32x8 acc[4];
#pragma unroll
    for (int i = 0; i < 4; ++i) acc[i] = zero8();

    const int r  = tid >> 1;
    const int hb = (tid & 1) * 16;
    const unsigned lA[2] = {(unsigned)(size_t)&As[0][r * 32 + hb],
                            (unsigned)(size_t)&As[1][r * 32 + hb]};
    const unsigned lB[2] = {(unsigned)(size_t)&Bs[0][r * 32 + hb],
                            (unsigned)(size_t)&Bs[1][r * 32 + hb]};
    const int rowA = ((m0 + r) * Cd + hb) * 2;
    const int rowB = ((n0 + r) * Cd + hb) * 2;

    stage_tile(lA[0], lB[0], Ab, Wb, rowA, rowB);
    for (int i = 0; i < KSTEPS; ++i) {
        const int cur = i & 1;
        if (i + 1 < KSTEPS) {
            stage_tile(lA[cur ^ 1], lB[cur ^ 1], Ab, Wb,
                       rowA + (i + 1) * 64, rowB + (i + 1) * 64);
            wait_async4();
        } else {
            wait_async0();
        }
        __syncthreads();

        Frag a;
        load_frag_rows(a, As[cur], wave * 16, 32, lane);
#pragma unroll
        for (int nt = 0; nt < 4; ++nt) {
            Frag b;
            load_frag_rows(b, Bs[cur], nt * 16, 32, lane);
            acc[nt] = wmma_bf16(a, b, acc[nt]);
        }
        __syncthreads();
    }

#pragma unroll
    for (int nt = 0; nt < 4; ++nt) {
#pragma unroll
        for (int rr = 0; rr < 8; ++rr) {
            int m  = wave * 16 + rr + ((lane >> 4) << 3);
            int o  = n0 + nt * 16 + (lane & 15);
            int gr = m0 + m;
            out[(size_t)gr * Cd + o] = acc[nt][rr] + proj_b[o];
        }
    }
}

// ---------- launch ----------
extern "C" void kernel_launch(void* const* d_in, const int* in_sizes, int n_in,
                              void* d_out, int out_size, void* d_ws, size_t ws_size,
                              hipStream_t stream) {
    const float* x        = (const float*)d_in[0];   // [2048,64,512]
    const float* mask     = (const float*)d_in[1];   // [64,64,64]
    const float* qkv_w    = (const float*)d_in[2];   // [1536,512]
    const float* q_bias   = (const float*)d_in[3];   // [512]
    const float* v_bias   = (const float*)d_in[4];   // [512]
    const float* lscale   = (const float*)d_in[5];   // [16,1,1]
    const float* cpb_w1   = (const float*)d_in[6];   // [512,2]
    const float* cpb_b1   = (const float*)d_in[7];   // [512]
    const float* cpb_w2   = (const float*)d_in[8];   // [16,512]
    const float* proj_w   = (const float*)d_in[9];   // [512,512]
    const float* proj_b   = (const float*)d_in[10];  // [512]
    const float* rel_tab  = (const float*)d_in[11];  // [1,15,15,2]
    const int*   rel_idx  = (const int*)d_in[12];    // [64,64]
    float*       out      = (float*)d_out;

    char*  w   = (char*)d_ws;
    size_t off = 0;
    auto bump = [&](size_t bytes) -> void* {
        void* p = w + off;
        off += (bytes + 255) & ~(size_t)255;
        return p;
    };
    unsigned short* xb    = (unsigned short*)bump((size_t)ROWS * Cd * 2);        // 134 MB
    unsigned short* wqkv  = (unsigned short*)bump((size_t)3 * Cd * Cd * 2);      // 1.6 MB
    unsigned short* wproj = (unsigned short*)bump((size_t)Cd * Cd * 2);          // 0.5 MB
    unsigned short* qkvb  = (unsigned short*)bump((size_t)3 * Bw * Hh * Nt * Dd * 2); // 403 MB
    unsigned short* ao    = (unsigned short*)bump((size_t)ROWS * Cd * 2);        // 134 MB
    float*          btab  = (float*)bump((size_t)225 * 16 * 4);
    float*          bfull = (float*)bump((size_t)3 * Cd * 4);
    float*          bcb   = (float*)bump((size_t)NW * Hh * Nt * Nt * 4);         // 16.8 MB

    k_cvt<<<4096, 256, 0, stream>>>(x, xb, (size_t)ROWS * Cd);
    k_cvt<<<1024, 256, 0, stream>>>(qkv_w, wqkv, (size_t)3 * Cd * Cd);
    k_cvt<<<512, 256, 0, stream>>>(proj_w, wproj, (size_t)Cd * Cd);
    k_qkvbias<<<6, 256, 0, stream>>>(q_bias, v_bias, bfull);

    k_cpb<<<225, 256, 0, stream>>>(rel_tab, cpb_w1, cpb_b1, cpb_w2, btab);
    k_bias<<<(NW * Hh * Nt * Nt + 255) / 256, 256, 0, stream>>>(mask, rel_idx, btab, bcb);

    k_gemm_qkv<<<dim3(ROWS / 64, (3 * Cd) / 64), 128, 0, stream>>>(xb, wqkv, bfull, qkvb);
    k_attn<<<dim3(Bw, Hh), 128, 0, stream>>>(qkvb, lscale, bcb, ao);
    k_gemm_proj<<<dim3(ROWS / 64, Cd / 64), 128, 0, stream>>>(ao, wproj, proj_b, out);
}
